// MoEExperts_9268539425012
// MI455X (gfx1250) — matrix operand order repaired and checked by reading, not compiled, not run on hardware
//
#include <hip/hip_runtime.h>
#include <hip/hip_bf16.h>

typedef __attribute__((ext_vector_type(16))) __bf16 v16bf;
typedef __attribute__((ext_vector_type(8)))  __bf16 v8bf;
typedef __attribute__((ext_vector_type(4)))  __bf16 v4bf;
typedef __attribute__((ext_vector_type(8)))  float  v8f;
typedef __attribute__((ext_vector_type(4)))  unsigned int v4u;
typedef __attribute__((ext_vector_type(4)))  int v4i;
typedef __attribute__((ext_vector_type(8)))  int v8i;

namespace {

constexpr int E_ = 8, T_ = 2048, H_ = 1024, I_ = 2048;
constexpr int BM = 128, BN = 64, BK = 32;
constexpr int PAD = 8;            // 8 bf16 = 16B pad: breaks bank conflicts, keeps 16B align
constexpr int LDA = BK + PAD;     // 40 bf16 = 80B row stride (16B aligned)

__device__ __forceinline__ v16bf combine16(v8bf lo, v8bf hi) {
  v16bf r;
#pragma unroll
  for (int i = 0; i < 8; ++i) { r[i] = lo[i]; r[i + 8] = hi[i]; }
  return r;
}

// Gather a 16-element bf16 fragment from LDS: 8 contiguous + 8 contiguous at hi_off.
__device__ __forceinline__ v16bf load_frag(const __bf16* p, int hi_off) {
  v8bf lo = *(const v8bf*)p;
  v8bf hi = *(const v8bf*)(p + hi_off);
  return combine16(lo, hi);
}

__device__ __forceinline__ int expert_start(const int* bs, int e) {
  int s = 0;
  for (int i = 0; i < e; ++i) s += bs[i];
  return s;
}

__device__ __forceinline__ v4bf cvt4(float4 v) {
  v4bf t;
  t[0] = (__bf16)v.x; t[1] = (__bf16)v.y; t[2] = (__bf16)v.z; t[3] = (__bf16)v.w;
  return t;
}

// TDM: DMA a 2D bf16 tile (tile_cols x tile_rows) global->LDS.
// tensor_dim1 = valid_rows (rows beyond it read as zero -> ragged zero-fill).
// Padding: +16B after every 64B stored => LDS row pitch = 80B = LDA bf16.
__device__ __forceinline__ void tdm_load_tile_bf16(
    unsigned lds_off, unsigned long long gaddr, int valid_rows,
    int row_stride_elems, int tile_rows, int tile_cols) {
  v4u g0;
  g0.x = 1u;                                              // count=1, user desc
  g0.y = lds_off;                                         // lds_addr [63:32]
  g0.z = (unsigned)(gaddr & 0xFFFFFFFFull);               // global_addr lo
  g0.w = (unsigned)((gaddr >> 32) & 0x1FFFFFFull)         // global_addr [56:32]
         | (2u << 30);                                    // type=2 ("image")
  v8i g1;
  g1[0] = (1 << 16)      // data_size = 2B
        | (1 << 20)      // pad_enable
        | (3 << 22)      // pad_interval: every 2^(3+1)=16 DWORDs (64B)
        | (3 << 25);     // pad_amount: 4 DWORDs (16B)
  g1[1] = (tile_cols & 0xFFFF) << 16;                     // tensor_dim0 lo16
  g1[2] = ((unsigned)tile_cols >> 16)                     // tensor_dim0 hi16
        | ((valid_rows & 0xFFFF) << 16);                  // tensor_dim1 lo16
  g1[3] = (((unsigned)valid_rows >> 16) & 0xFFFF)         // tensor_dim1 hi16
        | ((tile_cols & 0xFFFF) << 16);                   // tile_dim0
  g1[4] = tile_rows & 0xFFFF;                             // tile_dim1 (dim2=0)
  g1[5] = row_stride_elems;                               // tensor_dim0_stride lo32
  g1[6] = 0;                                              // stride hi16 | dim1_stride lo16
  g1[7] = 0;
  v4i z4 = {0, 0, 0, 0};
#if defined(__clang_major__) && (__clang_major__ >= 23)
  v8i z8 = {0, 0, 0, 0, 0, 0, 0, 0};
  __builtin_amdgcn_tensor_load_to_lds(g0, g1, z4, z4, z8, 0);
#else
  __builtin_amdgcn_tensor_load_to_lds(g0, g1, z4, z4, 0);
#endif
}

} // namespace

// ---------------------------------------------------------------------------
// Stage A: gated[t,i] = silu(h @ w1) * (h @ w3)  (per-expert ragged GEMM pair)
// ---------------------------------------------------------------------------
__global__ __launch_bounds__(256) void moe_gateup_kernel(
    const float* __restrict__ hid, const float* __restrict__ w1,
    const float* __restrict__ w3, const int* __restrict__ bs,
    __bf16* __restrict__ gated) {
  __shared__ __bf16 sA[2][BM * LDA];
  __shared__ __bf16 sB1[2][BN * LDA];
  __shared__ __bf16 sB3[2][BN * LDA];

  const int e = blockIdx.z;
  const int start = expert_start(bs, e);
  const int rows = bs[e];
  const int m0 = blockIdx.y * BM;
  if (m0 >= rows) return;
  const int mval = min(BM, rows - m0);
  const int n0 = blockIdx.x * BN;

  const int tid = threadIdx.x;
  const int lane = tid & 31;
  const int w = tid >> 5;
  const int wm = (w & 3) * 32;   // 4 waves along M
  const int wn = (w >> 2) * 32;  // 2 waves along N

  // B gather coords: each thread owns column n, K-span [k0, k0+8)
  const int bn = tid & 63;
  const int k0 = (tid >> 6) << 3;

  v8f c1[2][2] = {}, c3[2][2] = {};

  const float* ap  = hid + (size_t)(start + m0) * H_;
  const float* wp1 = w1 + (size_t)e * H_ * I_ + n0 + bn;
  const float* wp3 = w3 + (size_t)e * H_ * I_ + n0 + bn;

  float4 aReg[4];
  float bReg1[8], bReg3[8];

  auto glbLoad = [&](int kb) {
#pragma unroll
    for (int i = 0; i < 4; ++i) {
      int idx = tid + 256 * i;      // float4 chunk id, 8 chunks per row
      int r = idx >> 3;
      int c = (idx & 7) << 2;
      aReg[i] = (r < mval) ? *(const float4*)(ap + (size_t)r * H_ + kb + c)
                           : make_float4(0.f, 0.f, 0.f, 0.f);
    }
#pragma unroll
    for (int j = 0; j < 8; ++j) {   // coalesced: lanes span consecutive n
      bReg1[j] = wp1[(size_t)(kb + k0 + j) * I_];
      bReg3[j] = wp3[(size_t)(kb + k0 + j) * I_];
    }
  };

  auto ldsStore = [&](int p) {
#pragma unroll
    for (int i = 0; i < 4; ++i) {
      int idx = tid + 256 * i;
      int r = idx >> 3;
      int c = (idx & 7) << 2;
      *(v4bf*)&sA[p][r * LDA + c] = cvt4(aReg[i]);   // ds_store_b64
    }
    v8bf t1, t3;
#pragma unroll
    for (int j = 0; j < 8; ++j) { t1[j] = (__bf16)bReg1[j]; t3[j] = (__bf16)bReg3[j]; }
    *(v8bf*)&sB1[p][bn * LDA + k0] = t1;             // ds_store_b128
    *(v8bf*)&sB3[p][bn * LDA + k0] = t3;
  };

  glbLoad(0);
  int p = 0;
  for (int kb = 0; kb < H_; kb += BK) {
    ldsStore(p);
    if (kb + BK < H_) glbLoad(kb + BK);   // next tile in flight during compute
    __syncthreads();

    // Per-lane fragment gathers, contiguous K order (k = 16*(lane/16) + s) in A and B.
    const int lrow = lane & 15;
    const int koffA = (lane >> 4) << 4;   // A: K {0..15} vs {16..31}
    const int koffB = (lane >> 4) << 4;   // B: K {0..15} vs {16..31}
    v16bf a[2], bb1[2], bb3[2];
#pragma unroll
    for (int f = 0; f < 2; ++f)
      a[f] = load_frag(&sA[p][(wm + f * 16 + lrow) * LDA + koffA], 8);
#pragma unroll
    for (int g = 0; g < 2; ++g) {
      bb1[g] = load_frag(&sB1[p][(wn + g * 16 + lrow) * LDA + koffB], 8);
      bb3[g] = load_frag(&sB3[p][(wn + g * 16 + lrow) * LDA + koffB], 8);
    }
#pragma unroll
    for (int f = 0; f < 2; ++f)
#pragma unroll
      for (int g = 0; g < 2; ++g) {
        c1[f][g] = __builtin_amdgcn_wmma_f32_16x16x32_bf16(
            false, a[f], false, bb1[g], (short)0, c1[f][g], false, false);
        c3[f][g] = __builtin_amdgcn_wmma_f32_16x16x32_bf16(
            false, a[f], false, bb3[g], (short)0, c3[f][g], false, false);
      }
    p ^= 1;
  }

  // Epilogue: SiLU-gate fuse, store bf16. C layout: VGPR r -> M = r (+8 if lane>=16).
  const int rl = (lane < 16) ? 0 : 8;
  const int cl = lane & 15;
#pragma unroll
  for (int f = 0; f < 2; ++f)
#pragma unroll
    for (int g = 0; g < 2; ++g) {
      const int rbase = m0 + wm + f * 16 + rl;
      const int cbase = n0 + wn + g * 16 + cl;
#pragma unroll
      for (int r = 0; r < 8; ++r) {
        int row = rbase + r;
        if (row < rows) {
          float x = c1[f][g][r];
          float u = c3[f][g][r];
          float s = x / (1.0f + __expf(-x));
          gated[(size_t)(start + row) * I_ + cbase] = (__bf16)(s * u);
        }
      }
    }
}

// ---------------------------------------------------------------------------
// Stage B: out[t,h] = gated_bf16 @ w2   (per-expert ragged GEMM, fp32 out)
// A tile arrives via Tensor Data Mover DMA; B tile via VALU-converted staging.
// ---------------------------------------------------------------------------
__global__ __launch_bounds__(256) void moe_down_kernel(
    const __bf16* __restrict__ gated, const float* __restrict__ w2,
    const int* __restrict__ bs, float* __restrict__ out) {
  __shared__ __bf16 sA[2][BM * LDA];
  __shared__ __bf16 sB[2][BN * LDA];

  const int e = blockIdx.z;
  const int start = expert_start(bs, e);
  const int rows = bs[e];
  const int m0 = blockIdx.y * BM;
  if (m0 >= rows) return;
  const int mval = min(BM, rows - m0);
  const int n0 = blockIdx.x * BN;

  const int tid = threadIdx.x;
  const int lane = tid & 31;
  const int w = tid >> 5;
  const int wm = (w & 3) * 32;
  const int wn = (w >> 2) * 32;

  const int bn = tid & 63;
  const int k0 = (tid >> 6) << 3;

  v8f acc[2][2] = {};

  const __bf16* ap = gated + (size_t)(start + m0) * I_;
  const float* wp = w2 + (size_t)e * I_ * H_ + n0 + bn;

  float bReg[8];

  auto glbLoadB = [&](int kb) {
#pragma unroll
    for (int j = 0; j < 8; ++j)       // coalesced column gather of w2
      bReg[j] = wp[(size_t)(kb + k0 + j) * H_];
  };

  auto ldsStoreB = [&](int p) {
    v8bf t;
#pragma unroll
    for (int j = 0; j < 8; ++j) t[j] = (__bf16)bReg[j];
    *(v8bf*)&sB[p][bn * LDA + k0] = t;         // ds_store_b128
  };

  // Wave 0 issues the whole 128x32 bf16 A-tile as one TDM descriptor.
  auto issueTDM = [&](int p, int kb) {
    if (w == 0) {
      unsigned lds_off = (unsigned)(size_t)(void*)&sA[p][0];
      unsigned long long ga = (unsigned long long)(size_t)(const void*)(ap + kb);
      tdm_load_tile_bf16(lds_off, ga, mval, I_, BM, BK);
    }
  };

  issueTDM(0, 0);
  glbLoadB(0);
  int p = 0;
  for (int kb = 0; kb < I_; kb += BK) {
    ldsStoreB(p);
    if (kb + BK < I_) glbLoadB(kb + BK);
    if (w == 0) __builtin_amdgcn_s_wait_tensorcnt(0);  // A tile for kb landed
    __syncthreads();
    // Safe: every wave consumed buffer p^1 before reaching this barrier.
    if (kb + BK < I_) issueTDM(p ^ 1, kb + BK);        // DMA overlaps compute

    const int lrow = lane & 15;
    const int koffA = (lane >> 4) << 4;
    const int koffB = (lane >> 4) << 4;
    v16bf a[2], bb[2];
#pragma unroll
    for (int f = 0; f < 2; ++f)
      a[f] = load_frag(&sA[p][(wm + f * 16 + lrow) * LDA + koffA], 8);
#pragma unroll
    for (int g = 0; g < 2; ++g)
      bb[g] = load_frag(&sB[p][(wn + g * 16 + lrow) * LDA + koffB], 8);
#pragma unroll
    for (int f = 0; f < 2; ++f)
#pragma unroll
      for (int g = 0; g < 2; ++g)
        acc[f][g] = __builtin_amdgcn_wmma_f32_16x16x32_bf16(
            false, a[f], false, bb[g], (short)0, acc[f][g], false, false);
    p ^= 1;
  }

  const int rl = (lane < 16) ? 0 : 8;
  const int cl = lane & 15;
#pragma unroll
  for (int f = 0; f < 2; ++f)
#pragma unroll
    for (int g = 0; g < 2; ++g) {
      const int rbase = m0 + wm + f * 16 + rl;
      const int cbase = n0 + wn + g * 16 + cl;
#pragma unroll
      for (int r = 0; r < 8; ++r) {
        int row = rbase + r;
        if (row < rows)
          out[(size_t)(start + row) * H_ + cbase] = acc[f][g][r];
      }
    }
}

extern "C" void kernel_launch(void* const* d_in, const int* in_sizes, int n_in,
                              void* d_out, int out_size, void* d_ws, size_t ws_size,
                              hipStream_t stream) {
  (void)in_sizes; (void)n_in; (void)out_size; (void)ws_size;
  const float* hid = (const float*)d_in[0];   // hiddens   [T,H]
  const float* w1  = (const float*)d_in[1];   // w1_weight [E,H,I]
  const float* w2  = (const float*)d_in[2];   // w2_weight [E,I,H]
  const float* w3  = (const float*)d_in[3];   // w3_weight [E,H,I]
  const int*   bs  = (const int*)d_in[4];     // batch_sizes [E]

  __bf16* gated = (__bf16*)d_ws;              // T*I bf16 = 8 MB scratch

  dim3 blk(256);
  dim3 gA(I_ / BN, (T_ + BM - 1) / BM, E_);   // 32 x 16 x 8
  moe_gateup_kernel<<<gA, blk, 0, stream>>>(hid, w1, w3, bs, gated);

  dim3 gB(H_ / BN, (T_ + BM - 1) / BM, E_);   // 16 x 16 x 8
  moe_down_kernel<<<gB, blk, 0, stream>>>(gated, w2, bs, (float*)d_out);
}
